// GAGStateEncoder_54941221650673
// MI455X (gfx1250) — matrix operand order repaired and checked by reading, not compiled, not run on hardware
//
#include <hip/hip_runtime.h>
#include <cstdint>
#include <cstddef>

// ---------------------------------------------------------------------------
// GAGStateEncoder for MI455X (gfx1250): bf16 WMMA GEMM pipeline.
// ---------------------------------------------------------------------------

typedef __attribute__((ext_vector_type(16))) __bf16 v16bf;
typedef __attribute__((ext_vector_type(8)))  float  v8f;

constexpr int B_ = 32;
constexpr int N_ = 1024;
constexpr int E_ = 16384;
constexpr int D_ = 128;
constexpr int H_ = 8;
constexpr int KB_ = 256;   // K_BUILD

// d_out layout (floats): [h_att | state_value | node_mask | stage]
constexpr size_t OUT_HATT  = 0;
constexpr size_t OUT_SV    = (size_t)B_ * N_ * D_;          // 4194304
constexpr size_t OUT_MASK  = OUT_SV + (size_t)B_ * 258;     // +8256
constexpr size_t OUT_STAGE = OUT_MASK + (size_t)B_ * N_;    // +32768

union ABits { uint4 u[2]; v16bf v; };

static __device__ __forceinline__ v16bf load_frag16(const __bf16* base, int o0, int o1) {
  ABits t;
  t.u[0] = *reinterpret_cast<const uint4*>(base + o0);
  t.u[1] = *reinterpret_cast<const uint4*>(base + o1);
  return t.v;
}

// ---------------------------------------------------------------------------
// Generic wave-tile WMMA GEMM: C[M x 128] = act(A[M x K] @ Wt^T + bias) (+res)
//   A: bf16 row-major, lda elements.  Wt: bf16, (128 x K) row-major (W^T).
//   rowBlk/rowStride: logical->physical A-row remap (for hc[:, :256, :]).
// One wave = 16 rows x 128 cols; 8 accumulator tiles; K in steps of 32.
// ---------------------------------------------------------------------------
__global__ __launch_bounds__(128)
void wmma_gemm(const __bf16* __restrict__ A, int lda,
               const __bf16* __restrict__ Wt,
               const float* __restrict__ bias,
               float* __restrict__ outF, int ldoF,
               __bf16* __restrict__ outB, int ldoB, int colOffB,
               const float* __restrict__ res, int resMod, int resLd,
               int M, int K, int act,
               int rowBlk, int rowStride)
{
  const int wave = threadIdx.x >> 5;
  const int lane = threadIdx.x & 31;
  const int tile = blockIdx.x * 4 + wave;
  if (tile * 16 >= M) return;                 // wave-uniform; EXEC stays all-1s
  const int half = lane >> 4;
  const int mcol = lane & 15;

  v8f acc[8];
#pragma unroll
  for (int i = 0; i < 8; ++i) acc[i] = (v8f){0.f,0.f,0.f,0.f,0.f,0.f,0.f,0.f};

  const int lrow = tile * 16 + mcol;
  const int prow = (lrow / rowBlk) * rowStride + (lrow % rowBlk);
  const __bf16* aptr = A + (size_t)prow * lda;

  const int kts = K >> 5;
  for (int kt = 0; kt < kts; ++kt) {
    const int b0 = kt * 32 + (half ? 16 : 0);
    const int b1 = kt * 32 + (half ? 24 : 8);
    v16bf af = load_frag16(aptr, b0, b1);
    const int wko = kt * 32 + (half ? 16 : 0);
#pragma unroll
    for (int nt = 0; nt < 8; ++nt) {
      const __bf16* wp = Wt + (size_t)(nt * 16 + mcol) * K + wko;
      v16bf bf = *reinterpret_cast<const v16bf*>(wp);
      acc[nt] = __builtin_amdgcn_wmma_f32_16x16x32_bf16(
          false, af, false, bf, (short)0, acc[nt], false, false);
    }
  }

#pragma unroll
  for (int nt = 0; nt < 8; ++nt) {
    const int n = nt * 16 + mcol;
    const float bv = bias ? bias[n] : 0.f;
#pragma unroll
    for (int r = 0; r < 8; ++r) {
      const int m = (half ? 8 : 0) + r;
      const int row = tile * 16 + m;
      float v = acc[nt][r] + bv;
      if (act) v = tanhf(v);
      if (res) v += res[(size_t)(row % resMod) * resLd + n];
      if (outF) outF[(size_t)row * ldoF + n] = v;
      if (outB) outB[(size_t)row * ldoB + colOffB + n] = (__bf16)v;
    }
  }
}

// ---------------------------------------------------------------------------
// Edge MLP + scatter. Row rho in [0, 2*B*E): eg = rho>>1, p = rho&1.
// gather h[eidx[eg][p]] -> tanh(x@W1+b1) -> tanh(.@W2+b2) -> atomic-add to
// s[eidx[eg][p^1]].  Layer1 C-layout -> LDS -> layer2 A-layout, fenced with
// s_wait_dscnt (cross-lane handoff within the wave).
// ---------------------------------------------------------------------------
__global__ __launch_bounds__(128)
void edge_mlp_scatter(const __bf16* __restrict__ h_bf,
                      const int* __restrict__ eidx,
                      const __bf16* __restrict__ Wt1, const float* __restrict__ b1,
                      const __bf16* __restrict__ Wt2, const float* __restrict__ b2,
                      float* __restrict__ sacc)
{
  __shared__ __align__(16) __bf16 mid[4][16][D_];
  __shared__ int dsts[4][16];

  const int wave = threadIdx.x >> 5;
  const int lane = threadIdx.x & 31;
  const int tile = blockIdx.x * 4 + wave;
  const int half = lane >> 4;
  const int mcol = lane & 15;

  const int rho = tile * 16 + mcol;
  const int eg  = rho >> 1;
  const int p   = rho & 1;
  const int b   = eg >> 14;                 // eg / E_
  const int* ep = eidx + (size_t)eg * 2;
  const int sn = ep[p];
  const int dn = ep[p ^ 1];
  if (lane < 16) dsts[wave][mcol] = dn;

  const __bf16* aptr = h_bf + ((size_t)(b * N_ + sn)) * D_;

  v8f acc[8];
#pragma unroll
  for (int i = 0; i < 8; ++i) acc[i] = (v8f){0.f,0.f,0.f,0.f,0.f,0.f,0.f,0.f};

  // ---- layer 1: A from global gathered rows ----
#pragma unroll
  for (int kt = 0; kt < 4; ++kt) {
    const int b0 = kt * 32 + (half ? 16 : 0);
    const int b1o = kt * 32 + (half ? 24 : 8);
    v16bf af = load_frag16(aptr, b0, b1o);
    const int wko = kt * 32 + (half ? 16 : 0);
#pragma unroll
    for (int nt = 0; nt < 8; ++nt) {
      v16bf bf = *reinterpret_cast<const v16bf*>(Wt1 + (size_t)(nt * 16 + mcol) * D_ + wko);
      acc[nt] = __builtin_amdgcn_wmma_f32_16x16x32_bf16(
          false, af, false, bf, (short)0, acc[nt], false, false);
    }
  }

  // ---- bias+tanh, re-lay C layout into row-major LDS tile ----
#pragma unroll
  for (int nt = 0; nt < 8; ++nt) {
    const int n = nt * 16 + mcol;
    const float bv = b1[n];
#pragma unroll
    for (int r = 0; r < 8; ++r) {
      const int m = (half ? 8 : 0) + r;
      mid[wave][m][n] = (__bf16)tanhf(acc[nt][r] + bv);
    }
  }
  asm volatile("s_wait_dscnt 0x0" ::: "memory");   // fence LDS stores vs loads

  // ---- layer 2: A from LDS ----
  v8f acc2[8];
#pragma unroll
  for (int i = 0; i < 8; ++i) acc2[i] = (v8f){0.f,0.f,0.f,0.f,0.f,0.f,0.f,0.f};
  const __bf16* lptr = &mid[wave][mcol][0];
#pragma unroll
  for (int kt = 0; kt < 4; ++kt) {
    const int b0 = kt * 32 + (half ? 16 : 0);
    const int b1o = kt * 32 + (half ? 24 : 8);
    v16bf af = load_frag16(lptr, b0, b1o);
    const int wko = kt * 32 + (half ? 16 : 0);
#pragma unroll
    for (int nt = 0; nt < 8; ++nt) {
      v16bf bf = *reinterpret_cast<const v16bf*>(Wt2 + (size_t)(nt * 16 + mcol) * D_ + wko);
      acc2[nt] = __builtin_amdgcn_wmma_f32_16x16x32_bf16(
          false, af, false, bf, (short)0, acc2[nt], false, false);
    }
  }

  // ---- bias+tanh, scatter-add to destination nodes ----
  float* sbase = sacc + (size_t)b * N_ * D_;
#pragma unroll
  for (int nt = 0; nt < 8; ++nt) {
    const int n = nt * 16 + mcol;
    const float bv = b2[n];
#pragma unroll
    for (int r = 0; r < 8; ++r) {
      const int m = (half ? 8 : 0) + r;
      const int dm = dsts[wave][m];
      unsafeAtomicAdd(sbase + (size_t)dm * D_ + n, tanhf(acc2[nt][r] + bv));
    }
  }
}

// ---------------------------------------------------------------------------
// Small helper kernels
// ---------------------------------------------------------------------------
__global__ void tr_kernel(const float* __restrict__ src, __bf16* __restrict__ dst,
                          int K, int N, int srcLd, int colOff) {
  int i = blockIdx.x * blockDim.x + threadIdx.x;
  if (i >= K * N) return;
  int k = i / N, n = i % N;
  dst[(size_t)n * K + k] = (__bf16)src[(size_t)k * srcLd + colOff + n];
}

__global__ void f2b_kernel(const float* __restrict__ in, __bf16* __restrict__ out, size_t n) {
  size_t i = (size_t)blockIdx.x * blockDim.x + threadIdx.x;
  if (i < n) out[i] = (__bf16)in[i];
}

// out[(i/128)*ostride + ooff + i%128] = bf16(in[i])
__global__ void bf16_scatter_kernel(const float* __restrict__ in, __bf16* __restrict__ out,
                                    int ostride, int ooff, size_t n) {
  size_t i = (size_t)blockIdx.x * blockDim.x + threadIdx.x;
  if (i < n) out[(i >> 7) * ostride + ooff + (i & 127)] = (__bf16)in[i];
}

__global__ void deg_kernel(const int* __restrict__ ei, float* __restrict__ deg) {
  int i = blockIdx.x * blockDim.x + threadIdx.x;
  if (i >= B_ * E_) return;
  int b = i >> 14;                      // i / E_
  unsafeAtomicAdd(&deg[b * N_ + ei[2 * i]],     1.f);
  unsafeAtomicAdd(&deg[b * N_ + ei[2 * i + 1]], 1.f);
}

__global__ void hcomb_kernel(const float* __restrict__ s, const float* __restrict__ deg,
                             __bf16* __restrict__ comb, size_t n) {
  size_t i = (size_t)blockIdx.x * blockDim.x + threadIdx.x;
  if (i >= n) return;
  float v = s[i] / (deg[i >> 7] + 1e-6f);
  comb[(i >> 7) * 256 + 128 + (i & 127)] = (__bf16)v;
}

__global__ void combine_kernel(const float* __restrict__ gate, const float* __restrict__ upd,
                               const float* __restrict__ h, __bf16* __restrict__ hc,
                               int ooff, size_t n) {
  size_t i = (size_t)blockIdx.x * blockDim.x + threadIdx.x;
  if (i >= n) return;
  float g = gate[i];
  float v = g * upd[i] + (1.f - g) * h[i];
  hc[(i >> 7) * 256 + ooff + (i & 127)] = (__bf16)v;
}

__global__ void hnum_kernel(const float* __restrict__ num,
                            const float* __restrict__ W0, const float* __restrict__ b0,
                            const float* __restrict__ W1, const float* __restrict__ b1,
                            float* __restrict__ out) {
  __shared__ float hid[256];
  int b = blockIdx.x, t = threadIdx.x;
  const float* x = num + b * 64;
  float a = b0[t];
  for (int k = 0; k < 64; ++k) a += x[k] * W0[k * 256 + t];
  hid[t] = tanhf(a);
  __syncthreads();
  if (t < 128) {
    float a2 = b1[t];
    for (int k = 0; k < 256; ++k) a2 += hid[k] * W1[k * 128 + t];
    out[b * 128 + t] = tanhf(a2);
  }
}

// Flash-style attention: one thread = (b, q, head); 256 keys, hd=16.
__global__ void attn_kernel(const float* __restrict__ q2, const float* __restrict__ k2,
                            const float* __restrict__ v2, __bf16* __restrict__ ohb) {
  int idx = blockIdx.x * blockDim.x + threadIdx.x;
  if (idx >= B_ * N_ * H_) return;
  int hh = idx & 7;
  int qi = (idx >> 3) & (N_ - 1);
  int b  = idx >> 13;

  const float* qp = q2 + ((size_t)(b * N_ + qi) * D_ + hh * 16);
  float qv[16];
#pragma unroll
  for (int d = 0; d < 16; ++d) qv[d] = qp[d];

  const float* kb = k2 + (size_t)b * KB_ * D_ + hh * 16;
  const float* vb = v2 + (size_t)b * KB_ * D_ + hh * 16;

  float mx = -1e30f, l = 0.f;
  float o[16];
#pragma unroll
  for (int d = 0; d < 16; ++d) o[d] = 0.f;

  for (int ki = 0; ki < KB_; ++ki) {
    const float* kr = kb + (size_t)ki * D_;
    float s = 0.f;
#pragma unroll
    for (int d = 0; d < 16; ++d) s += qv[d] * kr[d];
    s *= 0.25f;                              // 1/sqrt(16)
    float nm = fmaxf(mx, s);
    float sc = __expf(mx - nm);
    float es = __expf(s - nm);
    l = l * sc + es;
    const float* vr = vb + (size_t)ki * D_;
#pragma unroll
    for (int d = 0; d < 16; ++d) o[d] = o[d] * sc + es * vr[d];
    mx = nm;
  }
  float inv = 1.f / l;
  __bf16* op = ohb + ((size_t)(b * N_ + qi) * D_ + hh * 16);
#pragma unroll
  for (int d = 0; d < 16; ++d) op[d] = (__bf16)(o[d] * inv);
}

__global__ void mean_kernel(const float* __restrict__ hatt, float* __restrict__ hmean) {
  int b = blockIdx.x, d = threadIdx.x;     // 128 threads
  const float* p = hatt + (size_t)b * N_ * D_ + d;
  float s = 0.f;
  for (int q = 0; q < N_; ++q) s += p[(size_t)q * D_];
  hmean[b * D_ + d] = s * (1.f / (float)N_);
}

__global__ void tail_kernel(float* __restrict__ out, const float* __restrict__ hnum,
                            const float* __restrict__ hmean, const float* __restrict__ stage) {
  int i = blockIdx.x * blockDim.x + threadIdx.x;
  if (i < B_ * 258) {
    int b = i / 258, j = i % 258;
    float v = (j < 128) ? hnum[b * 128 + j]
            : (j < 256) ? hmean[b * 128 + j - 128]
                        : stage[b * 2 + j - 256];
    out[OUT_SV + i] = v;
  }
  int mi = i - B_ * 258;
  if (mi >= 0 && mi < B_ * N_) out[OUT_MASK + mi] = 1.0f;
  int si = mi - B_ * N_;
  if (si >= 0 && si < B_ * 2) out[OUT_STAGE + si] = stage[si];
}

// ---------------------------------------------------------------------------
// Host launcher
// ---------------------------------------------------------------------------
static inline size_t align256(size_t x) { return (x + 255) & ~(size_t)255; }

extern "C" void kernel_launch(void* const* d_in, const int* in_sizes, int n_in,
                              void* d_out, int out_size, void* d_ws, size_t ws_size,
                              hipStream_t stream) {
  (void)in_sizes; (void)n_in; (void)out_size; (void)ws_size;

  const float* numerical = (const float*)d_in[0];
  const float* node_feat = (const float*)d_in[1];
  const float* stage     = (const float*)d_in[2];
  const float* W_num0    = (const float*)d_in[3];
  const float* b_num0    = (const float*)d_in[4];
  const float* W_num1    = (const float*)d_in[5];
  const float* b_num1    = (const float*)d_in[6];
  const float* W_node    = (const float*)d_in[7];
  const float* b_node    = (const float*)d_in[8];
  const float* pos_enc   = (const float*)d_in[9];
  const float* ew1       = (const float*)d_in[10];
  const float* eb1       = (const float*)d_in[11];
  const float* ew2       = (const float*)d_in[12];
  const float* eb2       = (const float*)d_in[13];
  const float* Wg        = (const float*)d_in[14];
  const float* bg        = (const float*)d_in[15];
  const float* Wu        = (const float*)d_in[16];
  const float* bu        = (const float*)d_in[17];
  const float* Wq1       = (const float*)d_in[18];
  const float* bq1       = (const float*)d_in[19];
  const float* Wk1       = (const float*)d_in[20];
  const float* bk1       = (const float*)d_in[21];
  const float* Wv1       = (const float*)d_in[22];
  const float* bv1       = (const float*)d_in[23];
  const float* W_in      = (const float*)d_in[24];
  const float* b_in      = (const float*)d_in[25];
  const float* W_out     = (const float*)d_in[26];
  const float* b_out     = (const float*)d_in[27];
  const int*   ei_dis    = (const int*)d_in[28];
  const int*   ei_od     = (const int*)d_in[29];

  float* out = (float*)d_out;

  // ---- workspace carve ----
  char* w = (char*)d_ws;
  size_t off = 0;
  auto carve = [&](size_t bytes) { void* p = w + off; off += align256(bytes); return p; };

  const size_t BN   = (size_t)B_ * N_;           // 32768
  const size_t BND  = BN * D_;                   // 4194304
  const size_t BKB  = (size_t)B_ * KB_;          // 8192
  const size_t BKBD = BKB * D_;

  __bf16* wt_node = (__bf16*)carve(32 * 128 * 2);
  __bf16* wt_e1a  = (__bf16*)carve(128 * 128 * 2);
  __bf16* wt_e1b  = (__bf16*)carve(128 * 128 * 2);
  __bf16* wt_e2a  = (__bf16*)carve(128 * 128 * 2);
  __bf16* wt_e2b  = (__bf16*)carve(128 * 128 * 2);
  __bf16* wt_g    = (__bf16*)carve(256 * 128 * 2);
  __bf16* wt_u    = (__bf16*)carve(256 * 128 * 2);
  __bf16* wt_q1   = (__bf16*)carve(256 * 128 * 2);
  __bf16* wt_k1   = (__bf16*)carve(256 * 128 * 2);
  __bf16* wt_v1   = (__bf16*)carve(256 * 128 * 2);
  __bf16* wt_iq   = (__bf16*)carve(128 * 128 * 2);
  __bf16* wt_ik   = (__bf16*)carve(128 * 128 * 2);
  __bf16* wt_iv   = (__bf16*)carve(128 * 128 * 2);
  __bf16* wt_out  = (__bf16*)carve(128 * 128 * 2);

  __bf16* nf_bf   = (__bf16*)carve(BN * 32 * 2);
  float*  h_f32   = (float*)carve(BND * 4);
  __bf16* h_bf    = (__bf16*)carve(BND * 2);
  float*  s_acc   = (float*)carve(BND * 4);
  float*  deg     = (float*)carve(BN * 4);
  __bf16* comb_bf = (__bf16*)carve(BN * 256 * 2);
  float*  gate    = (float*)carve(BND * 4);
  float*  upd     = (float*)carve(BND * 4);
  __bf16* hc_bf   = (__bf16*)carve(BN * 256 * 2);
  __bf16* q_bf    = (__bf16*)carve(BND * 2);
  __bf16* k_bf    = (__bf16*)carve(BKBD * 2);
  __bf16* v_bf    = (__bf16*)carve(BKBD * 2);
  float*  q2      = (float*)carve(BND * 4);
  float*  k2      = (float*)carve(BKBD * 4);
  float*  v2      = (float*)carve(BKBD * 4);
  __bf16* oh_bf   = (__bf16*)carve(BND * 2);
  float*  h_num   = (float*)carve(32 * 128 * 4);
  float*  hmean   = (float*)carve(32 * 128 * 4);

  auto ew = [](size_t n) { return (unsigned)((n + 255) / 256); };
  const int BIG = 1 << 30;

  // ---- weight transpose + bf16 conversion ----
  tr_kernel<<<ew(32 * 128), 256, 0, stream>>>(W_node, wt_node, 32, 128, 128, 0);
  tr_kernel<<<ew(128 * 128), 256, 0, stream>>>(ew1 + 2 * 16384, wt_e1a, 128, 128, 128, 0);
  tr_kernel<<<ew(128 * 128), 256, 0, stream>>>(ew1 + 3 * 16384, wt_e1b, 128, 128, 128, 0);
  tr_kernel<<<ew(128 * 128), 256, 0, stream>>>(ew2 + 2 * 16384, wt_e2a, 128, 128, 128, 0);
  tr_kernel<<<ew(128 * 128), 256, 0, stream>>>(ew2 + 3 * 16384, wt_e2b, 128, 128, 128, 0);
  tr_kernel<<<ew(256 * 128), 256, 0, stream>>>(Wg,  wt_g,  256, 128, 128, 0);
  tr_kernel<<<ew(256 * 128), 256, 0, stream>>>(Wu,  wt_u,  256, 128, 128, 0);
  tr_kernel<<<ew(256 * 128), 256, 0, stream>>>(Wq1, wt_q1, 256, 128, 128, 0);
  tr_kernel<<<ew(256 * 128), 256, 0, stream>>>(Wk1, wt_k1, 256, 128, 128, 0);
  tr_kernel<<<ew(256 * 128), 256, 0, stream>>>(Wv1, wt_v1, 256, 128, 128, 0);
  tr_kernel<<<ew(128 * 128), 256, 0, stream>>>(W_in, wt_iq, 128, 128, 384, 0);
  tr_kernel<<<ew(128 * 128), 256, 0, stream>>>(W_in, wt_ik, 128, 128, 384, 128);
  tr_kernel<<<ew(128 * 128), 256, 0, stream>>>(W_in, wt_iv, 128, 128, 384, 256);
  tr_kernel<<<ew(128 * 128), 256, 0, stream>>>(W_out, wt_out, 128, 128, 128, 0);

  // ---- numerical MLP (tiny) ----
  hnum_kernel<<<B_, 256, 0, stream>>>(numerical, W_num0, b_num0, W_num1, b_num1, h_num);

  // ---- node embed: h = tanh(nf @ W_node + b) + pos_enc ----
  f2b_kernel<<<ew(BN * 32), 256, 0, stream>>>(node_feat, nf_bf, BN * 32);
  wmma_gemm<<<(BN / 16 + 3) / 4, 128, 0, stream>>>(
      nf_bf, 32, wt_node, b_node, h_f32, 128, nullptr, 0, 0,
      pos_enc, N_, 128, (int)BN, 32, /*act=*/1, BIG, 0);
  f2b_kernel<<<ew(BND), 256, 0, stream>>>(h_f32, h_bf, BND);
  bf16_scatter_kernel<<<ew(BND), 256, 0, stream>>>(h_f32, comb_bf, 256, 0, BND);

  // ---- message passing (only last GCN layer matters: input is always h) ----
  for (int set = 0; set < 2; ++set) {
    const int*   eidx = set ? ei_od : ei_dis;
    const float* ebia = set ? eb2 : eb1;
    __bf16* w1 = set ? wt_e2a : wt_e1a;
    __bf16* w2 = set ? wt_e2b : wt_e1b;

    hipMemsetAsync(s_acc, 0, BND * 4, stream);
    hipMemsetAsync(deg, 0, BN * 4, stream);
    deg_kernel<<<ew((size_t)B_ * E_), 256, 0, stream>>>(eidx, deg);

    // 2*B*E rows / 16 per wave / 4 waves per block
    edge_mlp_scatter<<<(2 * B_ * E_ / 16) / 4, 128, 0, stream>>>(
        h_bf, eidx, w1, ebia + 2 * 128, w2, ebia + 3 * 128, s_acc);

    hcomb_kernel<<<ew(BND), 256, 0, stream>>>(s_acc, deg, comb_bf, BND);

    wmma_gemm<<<(BN / 16 + 3) / 4, 128, 0, stream>>>(
        comb_bf, 256, wt_g, bg, gate, 128, nullptr, 0, 0,
        nullptr, 1, 0, (int)BN, 256, 1, BIG, 0);
    wmma_gemm<<<(BN / 16 + 3) / 4, 128, 0, stream>>>(
        comb_bf, 256, wt_u, bu, upd, 128, nullptr, 0, 0,
        nullptr, 1, 0, (int)BN, 256, 1, BIG, 0);

    combine_kernel<<<ew(BND), 256, 0, stream>>>(gate, upd, h_f32, hc_bf, set * 128, BND);
  }

  // ---- Q/K/V first projections (K/V only over first 256 nodes per batch) ----
  wmma_gemm<<<(BN / 16 + 3) / 4, 128, 0, stream>>>(
      hc_bf, 256, wt_q1, bq1, nullptr, 0, q_bf, 128, 0,
      nullptr, 1, 0, (int)BN, 256, 0, BIG, 0);
  wmma_gemm<<<(BKB / 16 + 3) / 4, 128, 0, stream>>>(
      hc_bf, 256, wt_k1, bk1, nullptr, 0, k_bf, 128, 0,
      nullptr, 1, 0, (int)BKB, 256, 0, /*rowBlk=*/KB_, /*rowStride=*/N_);
  wmma_gemm<<<(BKB / 16 + 3) / 4, 128, 0, stream>>>(
      hc_bf, 256, wt_v1, bv1, nullptr, 0, v_bf, 128, 0,
      nullptr, 1, 0, (int)BKB, 256, 0, KB_, N_);

  // ---- W_in projections ----
  wmma_gemm<<<(BN / 16 + 3) / 4, 128, 0, stream>>>(
      q_bf, 128, wt_iq, b_in + 0, q2, 128, nullptr, 0, 0,
      nullptr, 1, 0, (int)BN, 128, 0, BIG, 0);
  wmma_gemm<<<(BKB / 16 + 3) / 4, 128, 0, stream>>>(
      k_bf, 128, wt_ik, b_in + 128, k2, 128, nullptr, 0, 0,
      nullptr, 1, 0, (int)BKB, 128, 0, BIG, 0);
  wmma_gemm<<<(BKB / 16 + 3) / 4, 128, 0, stream>>>(
      v_bf, 128, wt_iv, b_in + 256, v2, 128, nullptr, 0, 0,
      nullptr, 1, 0, (int)BKB, 128, 0, BIG, 0);

  // ---- attention ----
  attn_kernel<<<ew((size_t)B_ * N_ * H_), 256, 0, stream>>>(q2, k2, v2, oh_bf);

  // ---- output projection straight into d_out ----
  wmma_gemm<<<(BN / 16 + 3) / 4, 128, 0, stream>>>(
      oh_bf, 128, wt_out, b_out, out + OUT_HATT, 128, nullptr, 0, 0,
      nullptr, 1, 0, (int)BN, 128, 0, BIG, 0);

  // ---- mean, state_value, mask, stage ----
  mean_kernel<<<B_, 128, 0, stream>>>(out + OUT_HATT, hmean);
  tail_kernel<<<ew((size_t)B_ * 258 + B_ * N_ + B_ * 2), 256, 0, stream>>>(
      out, h_num, hmean, stage);
}
